// PairwiseBatchedDistance_75041668596339
// MI455X (gfx1250) — matrix operand order repaired and checked by reading, not compiled, not run on hardware
//
#include <hip/hip_runtime.h>

// ---------------------------------------------------------------------------
// Pairwise squared L2 distance: out = max(||x||^2 + ||y||^2 - 2 X Y^T, 0)
// X: [8192, 768] f32, Y: [8192, 768] f32, out: [8192, 8192] f32
//
// Fast path (ws large enough):
//   1) prep kernel: one-shot f32 -> split-bf16 (hi/lo) arrays + row norms in ws
//   2) GEMM kernel: double-buffered async global->LDS (ASYNCcnt) tile copies,
//      3x v_wmma_f32_16x16x32_bf16 per K=32 chunk (hi*hi + hi*lo + lo*hi),
//      f32 accumulation, distance epilogue.
// Fallback (small ws): round-1 style on-the-fly conversion kernel.
// ---------------------------------------------------------------------------

typedef __attribute__((ext_vector_type(16))) __bf16 v16bf;
typedef __attribute__((ext_vector_type(8)))  float  v8f;
typedef __attribute__((ext_vector_type(4)))  int    v4i;

#define BDIM   8192
#define NDIM   8192
#define DDIM   768
#define TM     128
#define TN     128
#define TK     32
#define KPAD   40      // 40 bf16 = 80 B row stride -> conflict-free b128 LDS reads
#define NCHUNK (DDIM / TK)

#if defined(__AMDGCN__) && __has_builtin(__builtin_amdgcn_global_load_async_to_lds_b128)
#define PD_ASYNC 1
#else
#define PD_ASYNC 0
#endif

typedef __attribute__((address_space(1))) v4i gas_v4i;   // global int4
typedef __attribute__((address_space(3))) v4i las_v4i;   // LDS int4

union FragBF16 {
  uint4 q[2];
  v16bf v;
};

__device__ __forceinline__ unsigned short bf16_rne_bits(float f) {
  unsigned u = __builtin_bit_cast(unsigned, f);
  u += 0x7FFFu + ((u >> 16) & 1u);    // round-to-nearest-even
  return (unsigned short)(u >> 16);
}
__device__ __forceinline__ float bf16_bits_to_f32(unsigned short h) {
  unsigned u = ((unsigned)h) << 16;
  return __builtin_bit_cast(float, u);
}

__device__ __forceinline__ void pd_copy_b128(const unsigned short* g, unsigned short* l) {
#if PD_ASYNC
  __builtin_amdgcn_global_load_async_to_lds_b128((gas_v4i*)g, (las_v4i*)l, 0, 0);
#else
  *(uint4*)l = *(const uint4*)g;
#endif
}

__device__ __forceinline__ void pd_wait_async() {
#if defined(__AMDGCN__) && __has_builtin(__builtin_amdgcn_s_wait_asynccnt)
  __builtin_amdgcn_s_wait_asynccnt(0);
#elif PD_ASYNC
  asm volatile("s_wait_asynccnt 0" ::: "memory");
#endif
}

// ---------------------------------------------------------------------------
// Prep kernel: one wave32 per row. Writes hi/lo bf16 split of each row and
// the row norm (computed in full f32 precision).
// ---------------------------------------------------------------------------
__global__ __launch_bounds__(256)
void pd_prep_kernel(const float* __restrict__ X, const float* __restrict__ Y,
                    unsigned short* __restrict__ xhi, unsigned short* __restrict__ xlo,
                    unsigned short* __restrict__ yhi, unsigned short* __restrict__ ylo,
                    float* __restrict__ xn, float* __restrict__ yn) {
  int gwave = (blockIdx.x * blockDim.x + threadIdx.x) >> 5;
  int lane  = threadIdx.x & 31;
  const float* src;
  unsigned short *hi, *lo;
  float* nrm;
  int row;
  if (gwave < BDIM) { src = X; hi = xhi; lo = xlo; nrm = xn; row = gwave; }
  else              { src = Y; hi = yhi; lo = ylo; nrm = yn; row = gwave - BDIM; }

  const float4* p = (const float4*)(src + (size_t)row * DDIM);
  float s = 0.0f;
  #pragma unroll
  for (int i = 0; i < DDIM / 4 / 32; ++i) {        // 6 float4 per lane
    int idx4 = lane + i * 32;
    float4 v = p[idx4];
    float vs[4] = {v.x, v.y, v.z, v.w};
    s += v.x * v.x + v.y * v.y + v.z * v.z + v.w * v.w;
    unsigned short h[4], l[4];
    #pragma unroll
    for (int c = 0; c < 4; ++c) {
      h[c] = bf16_rne_bits(vs[c]);
      l[c] = bf16_rne_bits(vs[c] - bf16_bits_to_f32(h[c]));
    }
    size_t o = (size_t)row * DDIM + idx4 * 4;
    uint2 t;
    t.x = (unsigned)h[0] | ((unsigned)h[1] << 16);
    t.y = (unsigned)h[2] | ((unsigned)h[3] << 16);
    *(uint2*)&hi[o] = t;
    t.x = (unsigned)l[0] | ((unsigned)l[1] << 16);
    t.y = (unsigned)l[2] | ((unsigned)l[3] << 16);
    *(uint2*)&lo[o] = t;
  }
  #pragma unroll
  for (int off = 16; off > 0; off >>= 1) s += __shfl_xor(s, off, 32);
  if (lane == 0) nrm[row] = s;
}

// ---------------------------------------------------------------------------
// Fast GEMM: pre-split bf16 inputs, async double-buffered LDS pipeline.
// Workgroup 256 threads (8 waves), 128x128 tile. Wave w: 32x64 sub-tile
// (mi = w&3 -> rows mi*32.., ni = w>>2 -> cols ni*64..), acc[2][4] v8f.
// ---------------------------------------------------------------------------
__global__ __launch_bounds__(256)
void pd_wmma_pre_kernel(const unsigned short* __restrict__ xhi,
                        const unsigned short* __restrict__ xlo,
                        const unsigned short* __restrict__ yhi,
                        const unsigned short* __restrict__ ylo,
                        const float* __restrict__ xn, const float* __restrict__ yn,
                        float* __restrict__ Out) {
  // [buffer][array: Ahi,Alo,Bhi,Blo][tile]
  __shared__ unsigned short lds[2][4][TM * KPAD];

  const int tid     = threadIdx.x;
  const int lane    = tid & 31;
  const int wave    = tid >> 5;
  const int mi      = wave & 3;     // M sub-block (rows mi*32..mi*32+31)
  const int ni      = wave >> 2;    // N sub-block (cols ni*64..ni*64+63)
  const int halfsel = lane >> 4;
  const int l15     = lane & 15;
  const int tileM   = blockIdx.y * TM;
  const int tileN   = blockIdx.x * TN;

  const unsigned short* srcs[4] = {xhi, xlo, yhi, ylo};
  const int rowbase[4] = {tileM, tileM, tileN, tileN};

  v8f acc[2][4];
  #pragma unroll
  for (int s = 0; s < 2; ++s)
    #pragma unroll
    for (int j = 0; j < 4; ++j) {
      v8f z = {0.f, 0.f, 0.f, 0.f, 0.f, 0.f, 0.f, 0.f};
      acc[s][j] = z;
    }

  // issue the tile copy for K-chunk kc into LDS buffer `buf`
  auto issue = [&](int kc, int buf) {
    int k0 = kc * TK;
    #pragma unroll
    for (int a = 0; a < 4; ++a) {
      #pragma unroll
      for (int i = 0; i < 2; ++i) {
        int f   = tid + i * 256;   // b128 index 0..511 (each = 8 bf16)
        int row = f >> 2;          // 0..127
        int q   = f & 3;           // 16B chunk within the 32-wide K slab
        const unsigned short* g = srcs[a] +
            (size_t)(rowbase[a] + row) * DDIM + k0 + q * 8;
        unsigned short* l = &lds[buf][a][row * KPAD + q * 8];
        pd_copy_b128(g, l);
      }
    }
  };

  issue(0, 0);

  for (int kc = 0; kc < NCHUNK; ++kc) {
    int buf = kc & 1;
    pd_wait_async();          // my loads for this buffer are done
    __syncthreads();          // everyone's loads done; prev compute done
    if (kc + 1 < NCHUNK) issue(kc + 1, buf ^ 1);

    const unsigned short* Ah = lds[buf][0];
    const unsigned short* Al = lds[buf][1];
    const unsigned short* Bh = lds[buf][2];
    const unsigned short* Bl = lds[buf][3];

    // A fragments for the wave's two 16-row slabs (ISA 16x32 bf16 layout)
    FragBF16 aH[2], aL[2];
    #pragma unroll
    for (int s = 0; s < 2; ++s) {
      int m  = mi * 32 + s * 16 + l15;
      int kb = halfsel * 8;          // K blocks {kb..kb+7, kb+16..kb+23}
      const unsigned short* pH = &Ah[m * KPAD + kb];
      const unsigned short* pL = &Al[m * KPAD + kb];
      aH[s].q[0] = *(const uint4*)(pH);
      aH[s].q[1] = *(const uint4*)(pH + 16);
      aL[s].q[0] = *(const uint4*)(pL);
      aL[s].q[1] = *(const uint4*)(pL + 16);
    }

    #pragma unroll
    for (int j = 0; j < 4; ++j) {
      FragBF16 bH, bL;
      int n  = ni * 64 + j * 16 + l15;
      int kb = halfsel * 8;          // K blocks {kb..kb+7, kb+16..kb+23}
      const unsigned short* pH = &Bh[n * KPAD + kb];
      const unsigned short* pL = &Bl[n * KPAD + kb];
      bH.q[0] = *(const uint4*)(pH);
      bH.q[1] = *(const uint4*)(pH + 16);
      bL.q[0] = *(const uint4*)(pL);
      bL.q[1] = *(const uint4*)(pL + 16);
      #pragma unroll
      for (int s = 0; s < 2; ++s) {
        acc[s][j] = __builtin_amdgcn_wmma_f32_16x16x32_bf16(
            false, aH[s].v, false, bH.v, (short)0, acc[s][j], false, false);
        acc[s][j] = __builtin_amdgcn_wmma_f32_16x16x32_bf16(
            false, aH[s].v, false, bL.v, (short)0, acc[s][j], false, false);
        acc[s][j] = __builtin_amdgcn_wmma_f32_16x16x32_bf16(
            false, aL[s].v, false, bH.v, (short)0, acc[s][j], false, false);
      }
    }
    __syncthreads();          // all reads of this buffer done before overwrite
  }

  // ---- epilogue: d = max(x2[m] + y2[n] - 2*dot, 0) ----
  #pragma unroll
  for (int s = 0; s < 2; ++s) {
    float xv[8];
    #pragma unroll
    for (int v = 0; v < 8; ++v)
      xv[v] = xn[tileM + mi * 32 + s * 16 + halfsel * 8 + v];
    #pragma unroll
    for (int j = 0; j < 4; ++j) {
      float yv = yn[tileN + ni * 64 + j * 16 + l15];
      #pragma unroll
      for (int v = 0; v < 8; ++v) {
        int m = tileM + mi * 32 + s * 16 + halfsel * 8 + v;
        int n = tileN + ni * 64 + j * 16 + l15;
        float d = xv[v] + yv - 2.0f * acc[s][j][v];
        Out[(size_t)m * NDIM + n] = fmaxf(d, 0.0f);
      }
    }
  }
}

// ---------------------------------------------------------------------------
// Fallback: norms-only kernel + on-the-fly-conversion GEMM (round-1 style),
// used when ws is too small for the pre-split arrays.
// ---------------------------------------------------------------------------
__global__ __launch_bounds__(256)
void pd_norms_kernel(const float* __restrict__ X, const float* __restrict__ Y,
                     float* __restrict__ norms) {
  int gwave = (blockIdx.x * blockDim.x + threadIdx.x) >> 5;
  int lane  = threadIdx.x & 31;
  const float* src;
  float* dst;
  int row;
  if (gwave < BDIM) { src = X; dst = norms;        row = gwave; }
  else              { src = Y; dst = norms + BDIM; row = gwave - BDIM; }
  const float4* p = (const float4*)(src + (size_t)row * DDIM);
  float s = 0.0f;
  #pragma unroll
  for (int i = 0; i < DDIM / 4 / 32; ++i) {
    float4 v = p[lane + i * 32];
    s += v.x * v.x + v.y * v.y + v.z * v.z + v.w * v.w;
  }
  #pragma unroll
  for (int off = 16; off > 0; off >>= 1) s += __shfl_xor(s, off, 32);
  if (lane == 0) dst[row] = s;
}

__global__ __launch_bounds__(256)
void pd_wmma_fly_kernel(const float* __restrict__ X, const float* __restrict__ Y,
                        const float* __restrict__ norms, float* __restrict__ Out) {
  __shared__ unsigned short Ahi[TM * KPAD];
  __shared__ unsigned short Alo[TM * KPAD];
  __shared__ unsigned short Bhi[TN * KPAD];
  __shared__ unsigned short Blo[TN * KPAD];

  const int tid     = threadIdx.x;
  const int lane    = tid & 31;
  const int wave    = tid >> 5;
  const int halfsel = lane >> 4;
  const int l15     = lane & 15;
  const int tileM   = blockIdx.y * TM;
  const int tileN   = blockIdx.x * TN;

  v8f acc[8];
  #pragma unroll
  for (int j = 0; j < 8; ++j) {
    v8f z = {0.f, 0.f, 0.f, 0.f, 0.f, 0.f, 0.f, 0.f};
    acc[j] = z;
  }

  for (int k0 = 0; k0 < DDIM; k0 += TK) {
    #pragma unroll
    for (int i = 0; i < 4; ++i) {
      int f   = tid + i * 256;
      int row = f >> 3;
      int q   = f & 7;
      const float* xp = X + (size_t)(tileM + row) * DDIM + k0 + q * 4;
      const float* yp = Y + (size_t)(tileN + row) * DDIM + k0 + q * 4;
      float4 xv = *(const float4*)xp;
      float4 yv = *(const float4*)yp;
      float xs[4] = {xv.x, xv.y, xv.z, xv.w};
      float ys[4] = {yv.x, yv.y, yv.z, yv.w};
      unsigned short xh[4], xl[4], yh[4], yl[4];
      #pragma unroll
      for (int c = 0; c < 4; ++c) {
        xh[c] = bf16_rne_bits(xs[c]);
        xl[c] = bf16_rne_bits(xs[c] - bf16_bits_to_f32(xh[c]));
        yh[c] = bf16_rne_bits(ys[c]);
        yl[c] = bf16_rne_bits(ys[c] - bf16_bits_to_f32(yh[c]));
      }
      int o = row * KPAD + q * 4;
      uint2 t;
      t.x = (unsigned)xh[0] | ((unsigned)xh[1] << 16);
      t.y = (unsigned)xh[2] | ((unsigned)xh[3] << 16);
      *(uint2*)&Ahi[o] = t;
      t.x = (unsigned)xl[0] | ((unsigned)xl[1] << 16);
      t.y = (unsigned)xl[2] | ((unsigned)xl[3] << 16);
      *(uint2*)&Alo[o] = t;
      t.x = (unsigned)yh[0] | ((unsigned)yh[1] << 16);
      t.y = (unsigned)yh[2] | ((unsigned)yh[3] << 16);
      *(uint2*)&Bhi[o] = t;
      t.x = (unsigned)yl[0] | ((unsigned)yl[1] << 16);
      t.y = (unsigned)yl[2] | ((unsigned)yl[3] << 16);
      *(uint2*)&Blo[o] = t;
    }
    __syncthreads();

    FragBF16 aH, aL;
    {
      int m  = wave * 16 + l15;
      int kb = halfsel * 8;
      const unsigned short* pH = &Ahi[m * KPAD + kb];
      const unsigned short* pL = &Alo[m * KPAD + kb];
      aH.q[0] = *(const uint4*)(pH);
      aH.q[1] = *(const uint4*)(pH + 16);
      aL.q[0] = *(const uint4*)(pL);
      aL.q[1] = *(const uint4*)(pL + 16);
    }
    #pragma unroll
    for (int j = 0; j < 8; ++j) {
      FragBF16 bH, bL;
      int n  = j * 16 + l15;
      int kb = halfsel * 8;
      const unsigned short* pH = &Bhi[n * KPAD + kb];
      const unsigned short* pL = &Blo[n * KPAD + kb];
      bH.q[0] = *(const uint4*)(pH);
      bH.q[1] = *(const uint4*)(pH + 16);
      bL.q[0] = *(const uint4*)(pL);
      bL.q[1] = *(const uint4*)(pL + 16);
      acc[j] = __builtin_amdgcn_wmma_f32_16x16x32_bf16(
          false, aH.v, false, bH.v, (short)0, acc[j], false, false);
      acc[j] = __builtin_amdgcn_wmma_f32_16x16x32_bf16(
          false, aH.v, false, bL.v, (short)0, acc[j], false, false);
      acc[j] = __builtin_amdgcn_wmma_f32_16x16x32_bf16(
          false, aL.v, false, bH.v, (short)0, acc[j], false, false);
    }
    __syncthreads();
  }

  float xv[8];
  #pragma unroll
  for (int v = 0; v < 8; ++v)
    xv[v] = norms[tileM + wave * 16 + halfsel * 8 + v];
  #pragma unroll
  for (int j = 0; j < 8; ++j) {
    float yv = norms[BDIM + tileN + j * 16 + l15];
    #pragma unroll
    for (int v = 0; v < 8; ++v) {
      int m = tileM + wave * 16 + halfsel * 8 + v;
      int n = tileN + j * 16 + l15;
      float d = xv[v] + yv - 2.0f * acc[j][v];
      Out[(size_t)m * NDIM + n] = fmaxf(d, 0.0f);
    }
  }
}

// ---------------------------------------------------------------------------
extern "C" void kernel_launch(void* const* d_in, const int* in_sizes, int n_in,
                              void* d_out, int out_size, void* d_ws, size_t ws_size,
                              hipStream_t stream) {
  (void)in_sizes; (void)n_in; (void)out_size;
  const float* x = (const float*)d_in[0];
  const float* y = (const float*)d_in[1];
  float* out = (float*)d_out;

  const size_t elems   = (size_t)BDIM * DDIM;
  const size_t bfBytes = elems * sizeof(unsigned short);
  const size_t need    = 4 * bfBytes + 2 * (size_t)BDIM * sizeof(float);

  dim3 grid(NDIM / TN, BDIM / TM);   // 64 x 64 tiles

  if (ws_size >= need) {
    unsigned short* xhi = (unsigned short*)d_ws;
    unsigned short* xlo = xhi + elems;
    unsigned short* yhi = xlo + elems;
    unsigned short* ylo = yhi + elems;
    float* xn = (float*)(ylo + elems);
    float* yn = xn + BDIM;
    pd_prep_kernel<<<(2 * BDIM * 32) / 256, 256, 0, stream>>>(x, y, xhi, xlo, yhi, ylo, xn, yn);
    pd_wmma_pre_kernel<<<grid, 256, 0, stream>>>(xhi, xlo, yhi, ylo, xn, yn, out);
  } else {
    float* norms = (float*)d_ws;     // [0,8192): ||x||^2 ; [8192,16384): ||y||^2
    pd_norms_kernel<<<(2 * BDIM * 32) / 256, 256, 0, stream>>>(x, y, norms);
    pd_wmma_fly_kernel<<<grid, 256, 0, stream>>>(x, y, norms, out);
  }
}
